// GTN_MultiHeadAttention_86895778333444
// MI455X (gfx1250) — matrix operand order repaired and checked by reading, not compiled, not run on hardware
//
#include <hip/hip_runtime.h>

// ---------------------------------------------------------------------------
// MHA fwd, B=8,S=1024,D=512,H=8,Q=V=64 on gfx1250.
// All GEMMs use V_WMMA_F32_16X16X32_BF16 with bf16x3 error-compensated split
// (A=Ah+Al, B=Bh+Bl; C += Ah*Bh + Ah*Bl + Al*Bh; fp32 accumulate) giving
// ~fp32 accuracy at ~2.7x fewer WMMA issue slots than the fp32 WMMA path.
// d_out = [ out: 8*1024*512 fp32 | heatmap: 64*1024*1024 fp32 ].
// ---------------------------------------------------------------------------

typedef float v8f  __attribute__((ext_vector_type(8)));
typedef __bf16 v16bf __attribute__((ext_vector_type(16)));
typedef unsigned int v4u __attribute__((ext_vector_type(4)));
typedef unsigned int v8u __attribute__((ext_vector_type(8)));
typedef unsigned short u16;

static_assert(sizeof(v16bf) == 32, "v16bf must be 32B");

__device__ __forceinline__ v8f wmma_bf16(v16bf a, v16bf b, v8f c) {
  // (neg_a, A, neg_b, B, c_mod, C, reuse_a, reuse_b)  [codegen-confirmed]
  return __builtin_amdgcn_wmma_f32_16x16x32_bf16(false, a, false, b, (short)0,
                                                 c, false, false);
}

__device__ __forceinline__ v8f v8zero() {
  v8f z;
  #pragma unroll
  for (int i = 0; i < 8; ++i) z[i] = 0.0f;
  return z;
}

__device__ __forceinline__ u16 bf16_rn(float x) {
  unsigned int u = __float_as_uint(x);
  unsigned int r = u + 0x7FFFu + ((u >> 16) & 1u);
  return (u16)(r >> 16);
}
__device__ __forceinline__ float bf16_f(u16 h) {
  return __uint_as_float(((unsigned int)h) << 16);
}
__device__ __forceinline__ void split2(float x, u16& hi, u16& lo) {
  hi = bf16_rn(x);
  lo = bf16_rn(x - bf16_f(hi));
}

// Build a v16bf fragment from two 16B-aligned runs of 8 bf16 each.
__device__ __forceinline__ v16bf ldfrag(const u16* p0, const u16* p1) {
  v4u a = *(const v4u*)p0;
  v4u b = *(const v4u*)p1;
  v8u t = __builtin_shufflevector(a, b, 0, 1, 2, 3, 4, 5, 6, 7);
  return __builtin_bit_cast(v16bf, t);
}

// ---------------------------------------------------------------------------
// Elementwise fp32 -> (hi,lo) bf16 split.  4 elems/thread.
// ---------------------------------------------------------------------------
__global__ __launch_bounds__(256)
void split_plain(const float* __restrict__ src, u16* __restrict__ hi,
                 u16* __restrict__ lo) {
  const size_t i4 = (size_t)blockIdx.x * 256 + threadIdx.x;
  const float4 v = *(const float4*)(src + i4 * 4);
  u16 h[4], l[4];
  split2(v.x, h[0], l[0]); split2(v.y, h[1], l[1]);
  split2(v.z, h[2], l[2]); split2(v.w, h[3], l[3]);
  uint2 hp, lp;
  hp.x = h[0] | ((unsigned)h[1] << 16); hp.y = h[2] | ((unsigned)h[3] << 16);
  lp.x = l[0] | ((unsigned)l[1] << 16); lp.y = l[2] | ((unsigned)l[3] << 16);
  *(uint2*)(hi + i4 * 4) = hp;
  *(uint2*)(lo + i4 * 4) = lp;
}

// fp32 W[512][512] -> transposed bf16 split Wt[n][k].
__global__ __launch_bounds__(256)
void split_transpose512(const float* __restrict__ W, u16* __restrict__ th,
                        u16* __restrict__ tl) {
  const int idx = blockIdx.x * 256 + threadIdx.x;
  const int k = idx >> 9, n = idx & 511;
  u16 h, l;
  split2(W[(size_t)k * 512 + n], h, l);
  th[(size_t)n * 512 + k] = h;
  tl[(size_t)n * 512 + k] = l;
}

// ---------------------------------------------------------------------------
// bf16x3 GEMM: C[8192x512] = A[8192x512] * W[512x512] + bias.
// A given as split bf16 row-major; W as split bf16 TRANSPOSED [n][k].
// Block = 256 thr (8 waves); block tile 16 x 128; wave tile 16x16; K chunks
// of 32 -> 16 iters x 3 WMMA.  MODE 0: q/k store, split bf16 [n][s][64].
// MODE 1: v store, split bf16 transposed [n][64][1024].  MODE 2: fp32 [m][n].
// ---------------------------------------------------------------------------
template <int MODE>
__global__ __launch_bounds__(256)
void gemm_bf16x3(const u16* __restrict__ Ah, const u16* __restrict__ Al,
                 const u16* __restrict__ Wth, const u16* __restrict__ Wtl,
                 const float* __restrict__ bias, u16* __restrict__ dh,
                 u16* __restrict__ dl, float* __restrict__ dout) {
  __shared__ __align__(16) u16 sAh[16][520];
  __shared__ __align__(16) u16 sAl[16][520];

  const int m0 = blockIdx.x * 16;
  const int w  = threadIdx.x >> 5;
  const int n0 = blockIdx.y * 128 + w * 16;

  // Cooperative staging of A panel (hi+lo), coalesced b128.
  {
    const int t = threadIdx.x;
    #pragma unroll
    for (int i = 0; i < 4; ++i) {   // 1024 uint4 per plane / 256 thr
      const int f   = t + i * 256;
      const int row = f >> 6;        // 64 uint4 per row
      const int c8  = (f & 63) * 8;
      *(v4u*)&sAh[row][c8] = *(const v4u*)(Ah + (size_t)(m0 + row) * 512 + c8);
      *(v4u*)&sAl[row][c8] = *(const v4u*)(Al + (size_t)(m0 + row) * 512 + c8);
    }
  }
  __syncthreads();

  const int lane = threadIdx.x & 31;
  const int ll = lane & 15, lh = lane >> 4;
  const int col = n0 + ll;
  const u16* wh = Wth + (size_t)col * 512 + lh * 16;
  const u16* wl = Wtl + (size_t)col * 512 + lh * 16;

  v8f c = v8zero();
  #pragma unroll 4
  for (int k0 = 0; k0 < 512; k0 += 32) {
    // A fragment: contiguous 16 bf16 at k0+lh*16.
    v16bf ah = ldfrag(&sAh[ll][k0 + lh * 16], &sAh[ll][k0 + lh * 16 + 8]);
    v16bf al = ldfrag(&sAl[ll][k0 + lh * 16], &sAl[ll][k0 + lh * 16 + 8]);
    // B fragment: contiguous 16 bf16 at Wt[col][k0 + lh*16].
    v16bf bh = ldfrag(wh + k0, wh + k0 + 8);
    v16bf bl = ldfrag(wl + k0, wl + k0 + 8);
    c = wmma_bf16(ah, bh, c);
    c = wmma_bf16(ah, bl, c);
    c = wmma_bf16(al, bh, c);
  }

  const float bv = bias[col];
  #pragma unroll
  for (int r = 0; r < 8; ++r) {
    const int row = m0 + r + 8 * lh;
    const float val = c[r] + bv;
    if (MODE == 2) {
      dout[(size_t)row * 512 + col] = val;
    } else {
      const int h = col >> 6, cc = col & 63;
      const int bb = row >> 10, s = row & 1023;
      const int n = h * 8 + bb;
      u16 hi, lo;
      split2(val, hi, lo);
      const size_t idx = (MODE == 0)
          ? ((size_t)n * 1024 + s) * 64 + cc       // [n][s][64]
          : (size_t)n * 65536 + (size_t)cc * 1024 + s;  // [n][64][1024]
      dh[idx] = hi;
      dl[idx] = lo;
    }
  }
}

// ---------------------------------------------------------------------------
// Fused attention, bf16x3 WMMA.  Block = (16-query tile, n=h*8+b), 8 waves.
// Each wave sweeps 4 key-tile PAIRS (32 keys = one K-chunk for attn@V).
// q-tile fragments hoisted to registers (kt-invariant).  Scores fp32 ->
// heatmap; exp in fp32; p split to bf16 hi/lo via per-wave LDS tiles.
// ---------------------------------------------------------------------------
__global__ __launch_bounds__(256)
void attn_bf16x3(const u16* __restrict__ Qh, const u16* __restrict__ Ql,
                 const u16* __restrict__ Kh, const u16* __restrict__ Kl,
                 const u16* __restrict__ Vth, const u16* __restrict__ Vtl,
                 float* __restrict__ heat, u16* __restrict__ wvh,
                 u16* __restrict__ wvl) {
  __shared__ __align__(16) u16 sqh[16][72];
  __shared__ __align__(16) u16 sql[16][72];
  __shared__ __align__(16) u16 pth[8][16][40];
  __shared__ __align__(16) u16 ptl[8][16][40];
  __shared__ float part[8][16][64];
  __shared__ float wsum[8][16];
  __shared__ float rs[16];

  const int qt = blockIdx.x;
  const int n  = blockIdx.y;
  const int hh = n >> 3, b = n & 7;
  const size_t nbase = (size_t)n * 65536;
  const int i0 = qt * 16;

  // Stage q tile hi+lo (128 uint4 per plane).
  {
    const int t = threadIdx.x;
    const int f = t & 127;
    const int row = f >> 3, c8 = (f & 7) * 8;
    const u16* src = (t < 128 ? Qh : Ql) + nbase + (size_t)(i0 + row) * 64 + c8;
    u16* dst = (t < 128 ? &sqh[row][c8] : &sql[row][c8]);
    *(v4u*)dst = *(const v4u*)src;
  }
  __syncthreads();

  const int w    = threadIdx.x >> 5;
  const int lane = threadIdx.x & 31;
  const int ll = lane & 15, lh = lane >> 4;

  // Hoisted q fragments: chunks k0=0 and k0=32, hi and lo.
  const v16bf ah0 = ldfrag(&sqh[ll][lh * 16],      &sqh[ll][lh * 16 + 8]);
  const v16bf ah1 = ldfrag(&sqh[ll][32 + lh * 16], &sqh[ll][40 + lh * 16]);
  const v16bf al0 = ldfrag(&sql[ll][lh * 16],      &sql[ll][lh * 16 + 8]);
  const v16bf al1 = ldfrag(&sql[ll][32 + lh * 16], &sql[ll][40 + lh * 16]);

  float acc[8];
  #pragma unroll
  for (int r = 0; r < 8; ++r) acc[r] = 0.0f;
  v8f cwv[4];
  #pragma unroll
  for (int t4 = 0; t4 < 4; ++t4) cwv[t4] = v8zero();

  for (int pi = 0; pi < 4; ++pi) {
    const int p  = w + 8 * pi;      // key-tile pair 0..31
    const int j0 = p * 32;

    // --- scores for the two 16-col halves of this pair ---
    #pragma unroll
    for (int jh = 0; jh < 2; ++jh) {
      const int j = j0 + jh * 16 + ll;
      const u16* kbh = Kh + nbase + (size_t)j * 64 + lh * 16;
      const u16* kbl = Kl + nbase + (size_t)j * 64 + lh * 16;
      v8f cs = v8zero();
      #pragma unroll
      for (int k0 = 0; k0 < 64; k0 += 32) {
        const v16bf qah = (k0 == 0) ? ah0 : ah1;
        const v16bf qal = (k0 == 0) ? al0 : al1;
        v16bf bh = ldfrag(kbh + k0, kbh + k0 + 8);
        v16bf bl = ldfrag(kbl + k0, kbl + k0 + 8);
        cs = wmma_bf16(qah, bh, cs);
        cs = wmma_bf16(qah, bl, cs);
        cs = wmma_bf16(qal, bh, cs);
      }
      // heatmap write + causal mask + exp + split into p-tile
      #pragma unroll
      for (int r = 0; r < 8; ++r) {
        const int iloc = r + 8 * lh;
        const int i = i0 + iloc;
        heat[(size_t)n * 1048576 + (size_t)i * 1024 + j] = cs[r];
        const float pv = (j <= i) ? __expf(cs[r]) : 0.0f;
        acc[r] += pv;
        u16 hi, lo;
        split2(pv, hi, lo);
        pth[w][iloc][jh * 16 + ll] = hi;
        ptl[w][iloc][jh * 16 + ll] = lo;
      }
    }

    // --- attn(16x32) @ v(32x64): one K=32 chunk, 4 col tiles ---
    const v16bf pah = ldfrag(&pth[w][ll][lh * 16], &pth[w][ll][lh * 16 + 8]);
    const v16bf pal = ldfrag(&ptl[w][ll][lh * 16], &ptl[w][ll][lh * 16 + 8]);
    #pragma unroll
    for (int t4 = 0; t4 < 4; ++t4) {
      const u16* vbh = Vth + nbase + (size_t)(t4 * 16 + ll) * 1024 + j0 + lh * 16;
      const u16* vbl = Vtl + nbase + (size_t)(t4 * 16 + ll) * 1024 + j0 + lh * 16;
      v16bf bh = ldfrag(vbh, vbh + 8);
      v16bf bl = ldfrag(vbl, vbl + 8);
      cwv[t4] = wmma_bf16(pah, bh, cwv[t4]);
      cwv[t4] = wmma_bf16(pah, bl, cwv[t4]);
      cwv[t4] = wmma_bf16(pal, bh, cwv[t4]);
    }
  }

  // Row-sum reduction (16-lane half-groups, wave32 shuffles).
  #pragma unroll
  for (int r = 0; r < 8; ++r) {
    float v = acc[r];
    v += __shfl_xor(v, 1);
    v += __shfl_xor(v, 2);
    v += __shfl_xor(v, 4);
    v += __shfl_xor(v, 8);
    if (ll == 0) wsum[w][r + 8 * lh] = v;
  }
  #pragma unroll
  for (int t4 = 0; t4 < 4; ++t4)
    #pragma unroll
    for (int r = 0; r < 8; ++r)
      part[w][r + 8 * lh][t4 * 16 + ll] = cwv[t4][r];

  __syncthreads();

  if (threadIdx.x < 16) {
    float s = 0.0f;
    for (int w2 = 0; w2 < 8; ++w2) s += wsum[w2][threadIdx.x];
    rs[threadIdx.x] = s;
  }
  __syncthreads();

  // Cross-wave sum, deferred 1/rowsum, split-store to merged wv (bf16 hi/lo).
  {
    const int idx = threadIdx.x * 4;
    const int row = idx >> 6;
    const int c0  = idx & 63;
    float vout[4] = {0.f, 0.f, 0.f, 0.f};
    for (int w2 = 0; w2 < 8; ++w2) {
      #pragma unroll
      for (int e = 0; e < 4; ++e) vout[e] += part[w2][row][c0 + e];
    }
    const float inv = 1.0f / rs[row];
    u16 h[4], l[4];
    #pragma unroll
    for (int e = 0; e < 4; ++e) split2(vout[e] * inv, h[e], l[e]);
    uint2 hp, lp;
    hp.x = h[0] | ((unsigned)h[1] << 16); hp.y = h[2] | ((unsigned)h[3] << 16);
    lp.x = l[0] | ((unsigned)l[1] << 16); lp.y = l[2] | ((unsigned)l[3] << 16);
    const size_t mrow = (size_t)(b * 1024 + i0 + row) * 512 + hh * 64 + c0;
    *(uint2*)(wvh + mrow) = hp;
    *(uint2*)(wvl + mrow) = lp;
  }
}

// ---------------------------------------------------------------------------
extern "C" void kernel_launch(void* const* d_in, const int* in_sizes, int n_in,
                              void* d_out, int out_size, void* d_ws, size_t ws_size,
                              hipStream_t stream) {
  (void)in_sizes; (void)n_in; (void)out_size; (void)ws_size;
  const float* x  = (const float*)d_in[0];
  const float* Wq = (const float*)d_in[1];
  const float* bq = (const float*)d_in[2];
  const float* Wk = (const float*)d_in[3];
  const float* bk = (const float*)d_in[4];
  const float* Wv = (const float*)d_in[5];
  const float* bv = (const float*)d_in[6];
  const float* Wo = (const float*)d_in[7];
  const float* bo = (const float*)d_in[8];

  float* out  = (float*)d_out;
  float* heat = out + 4194304;

  // Workspace carve-up (ushort units).  Big arrays are 4194304 elems each.
  u16* p = (u16*)d_ws;
  const size_t NB = 4194304;   // 64*1024*64 == 8192*512
  u16 *xh = p, *xl = xh + NB;
  u16 *qh = xl + NB, *ql = qh + NB, *kh = ql + NB, *kl = kh + NB;
  u16 *vth = kl + NB, *vtl = vth + NB;     // v transposed [n][64][1024]
  u16 *wvh = vtl + NB, *wvl = wvh + NB;    // merged wv [m][512]
  u16* wp = wvl + NB;
  const size_t WB = 262144;    // 512*512
  u16 *wqth = wp,          *wqtl = wqth + WB;
  u16 *wkth = wqtl + WB,   *wktl = wkth + WB;
  u16 *wvth2 = wktl + WB,  *wvtl2 = wvth2 + WB;
  u16 *woth = wvtl2 + WB,  *wotl = woth + WB;

  const dim3 blk(256);
  split_plain<<<dim3(4096), blk, 0, stream>>>(x, xh, xl);
  split_transpose512<<<dim3(1024), blk, 0, stream>>>(Wq, wqth, wqtl);
  split_transpose512<<<dim3(1024), blk, 0, stream>>>(Wk, wkth, wktl);
  split_transpose512<<<dim3(1024), blk, 0, stream>>>(Wv, wvth2, wvtl2);
  split_transpose512<<<dim3(1024), blk, 0, stream>>>(Wo, woth, wotl);

  gemm_bf16x3<0><<<dim3(512, 4), blk, 0, stream>>>(xh, xl, wqth, wqtl, bq, qh, ql, nullptr);
  gemm_bf16x3<0><<<dim3(512, 4), blk, 0, stream>>>(xh, xl, wkth, wktl, bk, kh, kl, nullptr);
  gemm_bf16x3<1><<<dim3(512, 4), blk, 0, stream>>>(xh, xl, wvth2, wvtl2, bv, vth, vtl, nullptr);

  attn_bf16x3<<<dim3(64, 64), blk, 0, stream>>>(qh, ql, kh, kl, vth, vtl, heat, wvh, wvl);

  gemm_bf16x3<2><<<dim3(512, 4), blk, 0, stream>>>(wvh, wvl, woth, wotl, bo, nullptr, nullptr, out);
}
